// DenoiseQNN_50122268344719
// MI455X (gfx1250) — matrix operand order repaired and checked by reading, not compiled, not run on hardware
//
#include <hip/hip_runtime.h>
#include <hip/hip_bf16.h>

typedef __attribute__((ext_vector_type(16))) _Float16 v16h;
typedef __attribute__((ext_vector_type(8)))  _Float16 v8h;
typedef __attribute__((ext_vector_type(8)))  float    v8f;
typedef __attribute__((ext_vector_type(4)))  float    v4f;

#define BATCH 16384
#define NOUT  3072
#define KDIM  128

// ---------------------------------------------------------------------------
// Kernel 0: convert w2 (3072 x 128 f32, row-major [n][k]) to f16 in workspace.
// ---------------------------------------------------------------------------
__global__ __launch_bounds__(256) void k_cvt_w2(const float* __restrict__ w2,
                                                _Float16* __restrict__ w2h) {
  int i = blockIdx.x * 256 + threadIdx.x;   // grid sized exactly: 3072*128/256
  w2h[i] = (_Float16)w2[i];
}

// ---------------------------------------------------------------------------
// Kernel 1: per batch row b (one block of 128 threads):
//   (a) border/total reductions of x[b] (3 channels of 32x32)
//   (b) closed-form conv+mean  -> pooled[0..3]
//   (c) 4-qubit statevector simulation (thread 0, registers only)
//   (d) h = relu(q @ w1^T + b1) -> f16 into hbuf (16384 x 128)
// ---------------------------------------------------------------------------
__global__ __launch_bounds__(128) void k_pool_q_fc1(
    const float* __restrict__ x,  const float* __restrict__ cw,
    const float* __restrict__ cb, const float* __restrict__ qw,
    const float* __restrict__ w1, const float* __restrict__ b1,
    _Float16* __restrict__ hbuf) {
  // per channel: [0]=T [1]=row0 [2]=row31 [3]=col0 [4]=col31
  //              [5]=x[0][0] [6]=x[0][31] [7]=x[31][0] [8]=x[31][31]
  __shared__ float chS[3][9];
  __shared__ float pooledS[4];
  __shared__ float qS[4];
  const int t = threadIdx.x;
  const int b = blockIdx.x;
  if (t < 27) (&chS[0][0])[t] = 0.0f;
  __syncthreads();

  const float* xb = x + (size_t)b * 3072;
  #pragma unroll
  for (int i = 0; i < 3; ++i) {
    const float* xc = xb + i * 1024;
    const int p0 = t * 8;                       // 128 threads * 8 = 1024 elems
    const v4f* xv = (const v4f*)(xc + p0);
    v4f va = __builtin_nontemporal_load(xv);
    v4f vb = __builtin_nontemporal_load(xv + 1);
    float vals[8] = {va.x, va.y, va.z, va.w, vb.x, vb.y, vb.z, vb.w};
    float T = 0.f, R0 = 0.f, R31 = 0.f, C0 = 0.f, C31 = 0.f;
    #pragma unroll
    for (int j = 0; j < 8; ++j) {
      const int p = p0 + j;                     // p = r*32 + c
      const float v = vals[j];
      T += v;
      if (p < 32)         R0  += v;             // r == 0
      if (p >= 992)       R31 += v;             // r == 31
      if ((p & 31) == 0)  C0  += v;             // c == 0
      if ((p & 31) == 31) C31 += v;             // c == 31
      if (p == 0)    chS[i][5] = v;
      if (p == 31)   chS[i][6] = v;
      if (p == 992)  chS[i][7] = v;
      if (p == 1023) chS[i][8] = v;
    }
    #pragma unroll
    for (int off = 16; off > 0; off >>= 1) {    // wave32 reduction
      T   += __shfl_down(T,   off, 32);
      R0  += __shfl_down(R0,  off, 32);
      R31 += __shfl_down(R31, off, 32);
      C0  += __shfl_down(C0,  off, 32);
      C31 += __shfl_down(C31, off, 32);
    }
    if ((t & 31) == 0) {
      atomicAdd(&chS[i][0], T);
      atomicAdd(&chS[i][1], R0);
      atomicAdd(&chS[i][2], R31);
      atomicAdd(&chS[i][3], C0);
      atomicAdd(&chS[i][4], C31);
    }
  }
  __syncthreads();

  // pooled[o] = cb[o] + (1/1024) * sum_{i,dy,dx} w[o,i,dy,dx] * S(i,dy,dx)
  // S(dy,dx) = T - exRow(dy) - exCol(dx) + corner(dy,dx)
  if (t < 4) {
    const int o = t;
    float sum = 0.f;
    for (int i = 0; i < 3; ++i) {
      const float T   = chS[i][0], R0  = chS[i][1], R31 = chS[i][2];
      const float C0  = chS[i][3], C31 = chS[i][4];
      const float x00 = chS[i][5], x0c = chS[i][6];
      const float xr0 = chS[i][7], xrc = chS[i][8];
      const float exR[3] = {R31, 0.f, R0};
      const float exC[3] = {C31, 0.f, C0};
      const float cor[3][3] = {{xrc, 0.f, xr0}, {0.f, 0.f, 0.f}, {x0c, 0.f, x00}};
      #pragma unroll
      for (int dy = 0; dy < 3; ++dy)
        #pragma unroll
        for (int dx = 0; dx < 3; ++dx)
          sum += cw[((o*3 + i)*3 + dy)*3 + dx] * (T - exR[dy] - exC[dx] + cor[dy][dx]);
    }
    pooledS[o] = cb[o] + sum * (1.0f / 1024.0f);
  }
  __syncthreads();

  // 4-qubit statevector sim, fully unrolled in registers (16 complex amps).
  if (t == 0) {
    float re[16], im[16];
    #pragma unroll
    for (int k = 0; k < 16; ++k) { re[k] = 0.f; im[k] = 0.f; }
    re[0] = 1.f;
    #pragma unroll
    for (int w = 0; w < 4; ++w) {               // RY(pooled[w]) on wire w
      const float th = 0.5f * pooledS[w];
      const float c = cosf(th), s = sinf(th);
      const int m = 1 << (3 - w);
      #pragma unroll
      for (int k = 0; k < 16; ++k) {
        if (k & m) continue;
        const int kb = k | m;
        const float ar = re[k], ai = im[k], br = re[kb], bi = im[kb];
        re[k]  = c*ar - s*br;  im[k]  = c*ai - s*bi;
        re[kb] = s*ar + c*br;  im[kb] = s*ai + c*bi;
      }
    }
    for (int l = 0; l < 3; ++l) {
      #pragma unroll
      for (int w = 0; w < 4; ++w) {             // RX(qw[l,w]) on wire w
        const float th = 0.5f * qw[l*4 + w];
        const float c = cosf(th), s = sinf(th);
        const int m = 1 << (3 - w);
        #pragma unroll
        for (int k = 0; k < 16; ++k) {          // new = [[c,-is],[-is,c]] * old
          if (k & m) continue;
          const int kb = k | m;
          const float ar = re[k], ai = im[k], br = re[kb], bi = im[kb];
          re[k]  = c*ar + s*bi;  im[k]  = c*ai - s*br;
          re[kb] = c*br + s*ai;  im[kb] = c*bi - s*ar;
        }
      }
      #pragma unroll
      for (int w = 0; w < 4; ++w) {             // CNOT ring: control w, target (w+1)%4
        const int mc = 1 << (3 - w);
        const int mt = 1 << (3 - ((w + 1) & 3));
        float nr[16], ni[16];
        #pragma unroll
        for (int k = 0; k < 16; ++k) {
          const int sidx = (k & mc) ? (k ^ mt) : k;
          nr[k] = re[sidx]; ni[k] = im[sidx];
        }
        #pragma unroll
        for (int k = 0; k < 16; ++k) { re[k] = nr[k]; im[k] = ni[k]; }
      }
    }
    #pragma unroll
    for (int o = 0; o < 4; ++o) {               // <Z_o> = sum probs * (1-2*bit)
      float acc = 0.f;
      #pragma unroll
      for (int k = 0; k < 16; ++k) {
        const float pr = re[k]*re[k] + im[k]*im[k];
        acc = ((k >> (3 - o)) & 1) ? (acc - pr) : (acc + pr);
      }
      qS[o] = acc;
    }
  }
  __syncthreads();

  // FC1 + ReLU, store h as f16 (WMMA A operand feed)
  float hv = b1[t];
  #pragma unroll
  for (int k = 0; k < 4; ++k) hv += qS[k] * w1[t*4 + k];
  hv = fmaxf(hv, 0.f);
  hbuf[(size_t)b * KDIM + t] = (_Float16)hv;
}

// ---------------------------------------------------------------------------
// Kernel 2: FC2 GEMM  out[M=16384, N=3072] = h[M,128] * w2^T + b2
// v_wmma_f32_16x16x32_f16, K = 128 -> 4 unrolled WMMA steps.
// Each wave computes a 32x32 tile (2x2 of 16x16). w2h/h are L2-resident;
// the 201 MB f32 output is streamed with non-temporal stores.
// A-fragment lane layout (16x32 f16): lane L holds row M=L&15;
//   K = (L>=16)*16 + e  -> two 16B contiguous loads per lane.
// B-fragment lane layout (32x16 f16): lane L holds col N=L&15;
//   K = (L>=16)*16 + e  -> one 32B contiguous load (w2 is (N,K) row-major).
// ---------------------------------------------------------------------------
__global__ __launch_bounds__(256) void k_fc2_wmma(
    const _Float16* __restrict__ hbuf, const _Float16* __restrict__ w2h,
    const float* __restrict__ b2, float* __restrict__ out) {
  const int lane = threadIdx.x & 31;
  const int wid  = blockIdx.x * 8 + (threadIdx.x >> 5);
  const int tx   = wid % (NOUT / 32);           // 96 N-tiles of 32
  const int ty   = wid / (NOUT / 32);           // 512 M-tiles of 32
  const int m0   = ty * 32;
  const int n0   = tx * 32;
  const int lh   = lane & 15;
  const int lg   = lane >> 4;

  const _Float16* arow0 = hbuf + (size_t)(m0 + lh)      * KDIM;
  const _Float16* arow1 = hbuf + (size_t)(m0 + 16 + lh) * KDIM;
  const _Float16* brow0 = w2h  + (size_t)(n0 + lh)      * KDIM;
  const _Float16* brow1 = w2h  + (size_t)(n0 + 16 + lh) * KDIM;

  v8f acc00 = {}, acc01 = {}, acc10 = {}, acc11 = {};

  #pragma unroll
  for (int kc = 0; kc < KDIM; kc += 32) {
    v16h a0, a1;
    {
      v8h lo0 = *(const v8h*)(arow0 + kc +     lg * 16);
      v8h hi0 = *(const v8h*)(arow0 + kc + 8 + lg * 16);
      v8h lo1 = *(const v8h*)(arow1 + kc +     lg * 16);
      v8h hi1 = *(const v8h*)(arow1 + kc + 8 + lg * 16);
      #pragma unroll
      for (int i = 0; i < 8; ++i) {
        a0[i] = lo0[i]; a0[8 + i] = hi0[i];
        a1[i] = lo1[i]; a1[8 + i] = hi1[i];
      }
    }
    v16h bb0 = *(const v16h*)(brow0 + kc + lg * 16);
    v16h bb1 = *(const v16h*)(brow1 + kc + lg * 16);
    acc00 = __builtin_amdgcn_wmma_f32_16x16x32_f16(false, a0, false, bb0, (short)0, acc00, false, false);
    acc01 = __builtin_amdgcn_wmma_f32_16x16x32_f16(false, a0, false, bb1, (short)0, acc01, false, false);
    acc10 = __builtin_amdgcn_wmma_f32_16x16x32_f16(false, a1, false, bb0, (short)0, acc10, false, false);
    acc11 = __builtin_amdgcn_wmma_f32_16x16x32_f16(false, a1, false, bb1, (short)0, acc11, false, false);
  }

  const float bz0 = b2[n0 + lh];
  const float bz1 = b2[n0 + 16 + lh];

  // C/D layout: VGPR v, lane L -> row M = v + (L>=16)*8, col N = L&15
  #pragma unroll
  for (int v = 0; v < 8; ++v) {
    const size_t r0 = (size_t)(m0 + v + lg * 8)      * NOUT;
    const size_t r1 = (size_t)(m0 + 16 + v + lg * 8) * NOUT;
    __builtin_nontemporal_store(acc00[v] + bz0, out + r0 + n0 + lh);
    __builtin_nontemporal_store(acc01[v] + bz1, out + r0 + n0 + 16 + lh);
    __builtin_nontemporal_store(acc10[v] + bz0, out + r1 + n0 + lh);
    __builtin_nontemporal_store(acc11[v] + bz1, out + r1 + n0 + 16 + lh);
  }
}

// ---------------------------------------------------------------------------
extern "C" void kernel_launch(void* const* d_in, const int* in_sizes, int n_in,
                              void* d_out, int out_size, void* d_ws, size_t ws_size,
                              hipStream_t stream) {
  const float* x      = (const float*)d_in[0];   // (16384,3,32,32)
  const float* conv_w = (const float*)d_in[1];   // (4,3,3,3)
  const float* conv_b = (const float*)d_in[2];   // (4,)
  const float* qwts   = (const float*)d_in[3];   // (3,4)
  const float* w1     = (const float*)d_in[4];   // (128,4)
  const float* b1     = (const float*)d_in[5];   // (128,)
  const float* w2     = (const float*)d_in[6];   // (3072,128)
  const float* b2     = (const float*)d_in[7];   // (3072,)
  float* out = (float*)d_out;                    // (16384,3072)

  _Float16* w2h  = (_Float16*)d_ws;                                   // 786432 B
  _Float16* hbuf = (_Float16*)((char*)d_ws + (size_t)NOUT * KDIM * 2); // 4 MB

  k_cvt_w2<<<(NOUT * KDIM) / 256, 256, 0, stream>>>(w2, w2h);
  k_pool_q_fc1<<<BATCH, 128, 0, stream>>>(x, conv_w, conv_b, qwts, w1, b1, hbuf);
  k_fc2_wmma<<<(BATCH / 32) * (NOUT / 32) / 8, 256, 0, stream>>>(hbuf, w2h, b2, out);
}
